// MultiheadEmbedding_24653112279252
// MI455X (gfx1250) — matrix operand order repaired and checked by reading, not compiled, not run on hardware
//
#include <hip/hip_runtime.h>
#include <hip/hip_bf16.h>

// ---------------------------------------------------------------------------
// MultiheadEmbedding fused kernel for gfx1250 (MI455X), f16 WMMA / f32 accum.
//   xp  = x @ W^T + b                      (proj_gemm, 32x64 wave tiles)
//   s   = sigmoid(scale * xh @ k_h)        (heads_kernel stage 1, WMMA)
//   out = s @ k_h^T                        (heads_kernel stage 2, WMMA, fused)
// proj_gemm is software-pipelined with two fragment register sets so the
// f32->f16 cvts for chunk k+32 fill the WMMA->VALU WAR hazard slots of the
// chunk-k WMMAs (removes compiler v_nop padding).
// heads_kernel: 32-row wave tile, B fragments reused by both 16-row subtiles;
// sigmoid scores bounce through LDS (stride 40: conflict-free + 16B aligned).
// Workspace: xp f16 (16MB) | emb_n f16 (4MB) | emb_t f16 (4MB)  = 24MB.
// ---------------------------------------------------------------------------

typedef __attribute__((ext_vector_type(16))) _Float16 v16h;
typedef __attribute__((ext_vector_type(8)))  _Float16 v8h;
typedef __attribute__((ext_vector_type(8)))  float    v8f;

#define B_ROWS 8192
#define F_DIM  1024
#define E_DIM  2048
#define NHEADS 16
#define DHEAD  64
#define LDS_STRIDE 40   // f16 elems per score row: 20 DWORDs -> conflict-free

static __device__ __forceinline__ v8f wmma_f16(v16h a, v16h b, v8f c) {
  // D(f32 16x16) = A(f16 16x32) * B(f16 32x16) + C
  return __builtin_amdgcn_wmma_f32_16x16x32_f16(
      /*neg_a=*/false, a, /*neg_b=*/false, b,
      /*c_mod=*/(short)0, c, /*reuse_a=*/false, /*reuse_b=*/false);
}

// ---------------------------------------------------------------------------
// Kernel 0: codebook f32 -> f16 in two layouts.
//   emb_n[f][e]      = f16(embedding[f][e])           (e contiguous)
//   emb_t[h][e][dd]  = f16(embedding[h*64+dd][e])     (dd contiguous)
// ---------------------------------------------------------------------------
__global__ void prep_embedding(const float* __restrict__ emb,
                               _Float16* __restrict__ emb_n,
                               _Float16* __restrict__ emb_t) {
  int tid = blockIdx.x * blockDim.x + threadIdx.x;   // 0 .. F*E-1
  int e = tid % E_DIM;
  int f = tid / E_DIM;
  int h = f / DHEAD, dd = f % DHEAD;
  _Float16 hv = (_Float16)emb[tid];
  emb_n[tid] = hv;
  emb_t[((size_t)h * E_DIM + e) * DHEAD + dd] = hv;
}

// ---------------------------------------------------------------------------
// Kernel 1: xp = f16(x @ W^T + b).   One wave computes a 32x64 tile.
// A frag (16x32 f16): lane holds row M=lane%16; 8-elem runs at K+(hi?8:0), +16.
// B frag (32x16 f16): lane holds col N=lane%16; 8-elem runs at K+(hi?8:0), +16 (as A).
// B[k][n] = W[n][k]  -> lane reads W row n contiguously. No transposes needed.
// ---------------------------------------------------------------------------
static __device__ __forceinline__ void load_frags(
    const float* __restrict__ ar0, const float* __restrict__ ar1,
    const float* const* __restrict__ br, int k, bool hiH,
    v16h& a0, v16h& a1, v16h b[4]) {
  const int ka = k + (hiH ? 8 : 0);
  v8f x00 = *(const v8f*)(ar0 + ka);
  v8f x01 = *(const v8f*)(ar0 + ka + 16);
  v8f x10 = *(const v8f*)(ar1 + ka);
  v8f x11 = *(const v8f*)(ar1 + ka + 16);
  #pragma unroll
  for (int j = 0; j < 8; ++j) {
    a0[j] = (_Float16)x00[j];  a0[j+8] = (_Float16)x01[j];
    a1[j] = (_Float16)x10[j];  a1[j+8] = (_Float16)x11[j];
  }
  const int kb = k + (hiH ? 8 : 0);
  #pragma unroll
  for (int j = 0; j < 4; ++j) {
    v8f w0 = *(const v8f*)(br[j] + kb);
    v8f w1 = *(const v8f*)(br[j] + kb + 16);
    #pragma unroll
    for (int q = 0; q < 8; ++q) { b[j][q] = (_Float16)w0[q]; b[j][q+8] = (_Float16)w1[q]; }
  }
}

__global__ void proj_gemm(const float* __restrict__ x, const float* __restrict__ w,
                          const float* __restrict__ bias, _Float16* __restrict__ xp) {
  const int lane   = threadIdx.x & 31;
  const int warp   = threadIdx.x >> 5;
  const int wid    = blockIdx.x * 8 + warp;   // 4096 wave tiles
  const int tn     = wid & 15;                // F/64 = 16 column tiles
  const int tm     = wid >> 4;                // B/32 = 256 row tiles
  const int m0     = tm * 32, n0 = tn * 64;
  const bool hiH   = lane >= 16;
  const int  lan   = lane & 15;

  v8f c[2][4] = {};
  const float* ar0 = x + (size_t)(m0 + lan) * F_DIM;
  const float* ar1 = ar0 + (size_t)16 * F_DIM;
  const float* br[4];
  #pragma unroll
  for (int j = 0; j < 4; ++j) br[j] = w + (size_t)(n0 + j * 16 + lan) * F_DIM;

  // Double-buffered K loop: fragments for the next 32-chunk are loaded and
  // converted while the current chunk's 8 WMMAs are in flight.
  v16h a0A, a1A, bA[4], a0B, a1B, bB[4];
  load_frags(ar0, ar1, br, 0, hiH, a0A, a1A, bA);

  for (int k = 0; k < F_DIM; k += 64) {
    load_frags(ar0, ar1, br, k + 32, hiH, a0B, a1B, bB);
    #pragma unroll
    for (int j = 0; j < 4; ++j) {
      c[0][j] = wmma_f16(a0A, bA[j], c[0][j]);
      c[1][j] = wmma_f16(a1A, bA[j], c[1][j]);
    }
    const int kn = (k + 64 < F_DIM) ? (k + 64) : (F_DIM - 32); // clamp: dummy
    load_frags(ar0, ar1, br, kn, hiH, a0A, a1A, bA);
    #pragma unroll
    for (int j = 0; j < 4; ++j) {
      c[0][j] = wmma_f16(a0B, bB[j], c[0][j]);
      c[1][j] = wmma_f16(a1B, bB[j], c[1][j]);
    }
  }

  // C layout: lane<16 -> N=lane, M=r;  lane>=16 -> N=lane-16, M=r+8.
  const int madd = hiH ? 8 : 0;
  #pragma unroll
  for (int j = 0; j < 4; ++j) {
    float bv = bias[n0 + j * 16 + lan];
    #pragma unroll
    for (int i = 0; i < 2; ++i) {
      #pragma unroll
      for (int r = 0; r < 8; ++r) {
        int row = m0 + i * 16 + r + madd;
        xp[(size_t)row * F_DIM + n0 + j * 16 + lan] = (_Float16)(c[i][j][r] + bv);
      }
    }
  }
}

// ---------------------------------------------------------------------------
// Kernel 2: fused per-head scores -> sigmoid -> weighted sum over codebook.
// One wave = (32-row tile, head).  Output 32x64 accumulators stay in VGPRs.
// E streamed in chunks of 32; B fragments (head/e-chunk only) reused by both
// 16-row subtiles.  Sigmoid scores bounce through 2.5KB of LDS per wave to
// convert C-layout -> A-fragment layout (wave-local ordering, no barrier).
// ---------------------------------------------------------------------------
__global__ void heads_kernel(const _Float16* __restrict__ xp,
                             const _Float16* __restrict__ emb_n,
                             const _Float16* __restrict__ emb_t,
                             float* __restrict__ out) {
  __shared__ _Float16 lds[8][32 * LDS_STRIDE];   // 8 waves x 2.5KB = 20KB
  const int  lane = threadIdx.x & 31;
  const int  warp = threadIdx.x >> 5;
  const bool hiH  = lane >= 16;
  const int  lan  = lane & 15;
  const int  h    = blockIdx.x >> 5;                        // 32 blocks / head
  const int  m0   = ((blockIdx.x & 31) * 8 + warp) * 32;    // 32-row tile
  const float scale = 0.125f;                               // 1/sqrt(64)

  // xh A-fragments, 2 subtiles x 16x64 f16: loaded once, reused for all of E.
  v16h a0[2], a1[2];
  #pragma unroll
  for (int i = 0; i < 2; ++i) {
    const _Float16* p = xp + (size_t)(m0 + i * 16 + lan) * F_DIM
                           + h * DHEAD + (hiH ? 16 : 0);
    v8h r0 = *(const v8h*)(p);        // K =  0.. 7 (+16 for hi lanes)
    v8h r1 = *(const v8h*)(p + 8);    // K =  8..15
    v8h r2 = *(const v8h*)(p + 32);   // K = 32..39
    v8h r3 = *(const v8h*)(p + 40);   // K = 40..47
    #pragma unroll
    for (int j = 0; j < 8; ++j) { a0[i][j] = r0[j]; a0[i][j+8] = r1[j];
                                  a1[i][j] = r2[j]; a1[i][j+8] = r3[j]; }
  }

  v8f acc[2][4] = {};
  _Float16* my = lds[warp];
  const _Float16* tbase = emb_t + (size_t)h * E_DIM * DHEAD;  // [e][dd]
  const _Float16* nbase = emb_n + (size_t)h * DHEAD * E_DIM;  // [dd][e]
  const int madd = hiH ? 8 : 0;

  for (int ec = 0; ec < E_DIM; ec += 32) {
    // ---- stage 1: scores S[32 x 32] = xh @ k_h[:, ec:ec+32] --------------
    v8f s[2][2] = {};
    #pragma unroll
    for (int t = 0; t < 2; ++t) {
      const _Float16* col = tbase + (size_t)(ec + t * 16 + lan) * DHEAD
                                  + (hiH ? 16 : 0);
      v16h blo = *(const v16h*)(col);        // K = dd  0..31 slice
      v16h bhi = *(const v16h*)(col + 32);   // K = dd 32..63 slice
      #pragma unroll
      for (int i = 0; i < 2; ++i) {
        s[i][t] = wmma_f16(a0[i], blo, s[i][t]);
        s[i][t] = wmma_f16(a1[i], bhi, s[i][t]);
      }
    }

    // ---- sigmoid, write f16 scores to LDS rows [32][32] (stride 40) -----
    #pragma unroll
    for (int i = 0; i < 2; ++i) {
      #pragma unroll
      for (int t = 0; t < 2; ++t) {
        #pragma unroll
        for (int r = 0; r < 8; ++r) {
          float sv = s[i][t][r] * scale;
          float wv = 1.0f / (1.0f + __expf(-sv));
          my[(i * 16 + r + madd) * LDS_STRIDE + t * 16 + lan] = (_Float16)wv;
        }
      }
    }
    asm volatile("s_wait_dscnt 0" ::: "memory");  // wave-local RAW on LDS

    // ---- reload as A-fragments (2 x 16x32) -------------------------------
    v16h wa[2];
    #pragma unroll
    for (int i = 0; i < 2; ++i) {
      const _Float16* p = my + (i * 16 + lan) * LDS_STRIDE + (hiH ? 16 : 0);
      v8h r0 = *(const v8h*)(p);
      v8h r1 = *(const v8h*)(p + 8);
      #pragma unroll
      for (int j = 0; j < 8; ++j) { wa[i][j] = r0[j]; wa[i][j+8] = r1[j]; }
    }

    // ---- stage 2: out[32 x 64] += S @ k_h[:, ec:ec+32]^T -----------------
    #pragma unroll
    for (int jd = 0; jd < 4; ++jd) {
      const _Float16* q = nbase + (size_t)(jd * 16 + lan) * E_DIM + ec
                                + (hiH ? 16 : 0);
      v16h bb = *(const v16h*)(q);           // B[k=e][n=dd], e contiguous
      acc[0][jd] = wmma_f16(wa[0], bb, acc[0][jd]);
      acc[1][jd] = wmma_f16(wa[1], bb, acc[1][jd]);
    }
  }

  // ---- store f32 output: out[b][h*64 + dd] -------------------------------
  #pragma unroll
  for (int jd = 0; jd < 4; ++jd) {
    int col = h * DHEAD + jd * 16 + lan;
    #pragma unroll
    for (int i = 0; i < 2; ++i) {
      #pragma unroll
      for (int r = 0; r < 8; ++r) {
        out[(size_t)(m0 + i * 16 + r + madd) * F_DIM + col] = acc[i][jd][r];
      }
    }
  }
}

// ---------------------------------------------------------------------------
extern "C" void kernel_launch(void* const* d_in, const int* in_sizes, int n_in,
                              void* d_out, int out_size, void* d_ws, size_t ws_size,
                              hipStream_t stream) {
  const float* x   = (const float*)d_in[0];   // [8192,1024]
  const float* pw  = (const float*)d_in[1];   // [1024,1024]
  const float* pb  = (const float*)d_in[2];   // [1024]
  const float* emb = (const float*)d_in[3];   // [1024,2048]
  float* out = (float*)d_out;                 // [8192,1024] f32

  _Float16* xp    = (_Float16*)d_ws;                                       // 16MB
  _Float16* emb_n = (_Float16*)((char*)d_ws + (size_t)B_ROWS * F_DIM * 2); // 4MB
  _Float16* emb_t = emb_n + (size_t)F_DIM * E_DIM;                         // 4MB

  prep_embedding<<<(F_DIM * E_DIM) / 256, 256, 0, stream>>>(emb, emb_n, emb_t);
  proj_gemm<<<(B_ROWS / 32) * (F_DIM / 64) / 8, 256, 0, stream>>>(x, pw, pb, xp);
  heads_kernel<<<NHEADS * 32, 256, 0, stream>>>(xp, emb_n, emb_t, out);
}
